// StateSlotBank_48378511622737
// MI455X (gfx1250) — matrix operand order repaired and checked by reading, not compiled, not run on hardware
//
#include <hip/hip_runtime.h>
#include <hip/hip_bf16.h>
#include <math.h>

// ---------------------------------------------------------------------------
// StateSlotBank forward for MI455X (gfx1250, wave32, WMMA).
// Phase 0: pre-swizzle W_in / W_out into WMMA B-fragment layout (bf16),
//          l2-normalize slot_keys.
// Phase 1: per (batch, 16-row tile): LN -> q GEMM (WMMA bf16) -> 64-slot
//          attention (f32 VALU) -> read -> out GEMM (WMMA bf16); also emits
//          per-chunk summary-derived quantities (wv, top3 idx, wv-gate term)
//          since they do NOT depend on the carried slot state.
// Phase 2: per-batch sequential scan (256 iters): gather 3 slots, one
//          reduction, sigmoid gate, blend; then LayerNorm -> new_slots.
// ---------------------------------------------------------------------------

typedef __attribute__((ext_vector_type(16))) __bf16 v16bf;
typedef __attribute__((ext_vector_type(8)))  __bf16 v8bf;
typedef __attribute__((ext_vector_type(8)))  float  v8f;

#define SHUF16(lo, hi) __builtin_shufflevector((lo), (hi), 0,1,2,3,4,5,6,7,8,9,10,11,12,13,14,15)

constexpr int Bb = 8, Ss = 4096, IN = 1024, SD = 256, NS = 64;
constexpr int NTILE = Ss / 16;      // 256 sequence tiles == 256 chunks

// ------------------------- phase 0: prep kernels ---------------------------

// W_in (1024 x 256, row major f32) -> bf16 B-fragments.
// Fragment element index: ((ntile*KSTEPS + kstep)*32 + lane)*16 + m
// lane = (n%16) + 16*((k%32) >= 16), m = k%16   (B layout: lanes 0-15 hold
// K=0..15, lanes 16-31 hold K=16..31 of the 32-deep step; lane picks column)
__global__ void prep_win(const float* __restrict__ W, __bf16* __restrict__ F) {
  int idx = blockIdx.x * 256 + threadIdx.x;          // 1024*256 elements
  int k = idx >> 8, n = idx & 255;
  int ntile = n >> 4, ks = k >> 5;
  int lane = (n & 15) + (((k & 31) >= 16) ? 16 : 0);
  int m = k & 15;
  F[(((ntile * 32 + ks) * 32 + lane) << 4) + m] = (__bf16)W[idx];
}

// W_out (256 x 1024, row major f32) -> bf16 B-fragments (64 ntiles, 8 ksteps)
__global__ void prep_wout(const float* __restrict__ W, __bf16* __restrict__ F) {
  int idx = blockIdx.x * 256 + threadIdx.x;          // 256*1024 elements
  int k = idx >> 10, n = idx & 1023;
  int ntile = n >> 4, ks = k >> 5;
  int lane = (n & 15) + (((k & 31) >= 16) ? 16 : 0);
  int m = k & 15;
  F[(((ntile * 8 + ks) * 32 + lane) << 4) + m] = (__bf16)W[idx];
}

// key_norms = l2norm(slot_keys) rows, f32
__global__ void prep_keyn(const float* __restrict__ K, float* __restrict__ O) {
  __shared__ float red[256];
  int row = blockIdx.x, tid = threadIdx.x;
  float v = K[row * 256 + tid];
  red[tid] = v * v;
  __syncthreads();
  for (int s = 128; s; s >>= 1) { if (tid < s) red[tid] += red[tid + s]; __syncthreads(); }
  float norm = fmaxf(sqrtf(red[0]), 1e-12f);
  O[row * 256 + tid] = v / norm;
}

// ------------------------- phase 1: fused forward --------------------------

__global__ __launch_bounds__(256)
void fused_fwd(const float* __restrict__ x,
               const float* __restrict__ slot_mem,
               const float* __restrict__ W_write,
               const float* __restrict__ W_gate,
               const float* __restrict__ ln_in_s,
               const float* __restrict__ ln_in_b,
               const __bf16* __restrict__ WinF,
               const __bf16* __restrict__ WoutF,
               const float* __restrict__ keyn,
               float* __restrict__ out,
               float* __restrict__ ws_wv,
               float* __restrict__ ws_zwv,
               int* __restrict__ ws_top)
{
  // 64KB LDS, phase-overlaid:
  //   region0 (16KB): q (16x256 f32)            -> read (16x256 f32)
  //   region1 (32KB): LN'd x (16x1024 bf16)     -> slots bf16 (64x256)
  //                                             -> read bf16 + summary + scratch
  //   region2 (16KB): scores/attn (16x4x64 f32)
  __shared__ __align__(32) char smem[65536];
  float*  shq  = (float*)smem;
  char*   r1   = smem + 16384;
  float*  shsc = (float*)(smem + 49152);
  __bf16* shA  = (__bf16*)r1;
  __bf16* shSl = (__bf16*)r1;
  __bf16* shRb = (__bf16*)r1;
  float*  shSum = (float*)(r1 + 8192);   // 256 f32
  float*  shMt  = (float*)(r1 + 9216);   // 64 f32
  float*  shRed = (float*)(r1 + 9472);   // 8 f32

  const int tid = threadIdx.x, wave = tid >> 5, lane = tid & 31;
  const int b = blockIdx.y, tile = blockIdx.x;
  const size_t rowbase = ((size_t)b * Ss + (size_t)tile * 16);

  // ---- step 1: LayerNorm(x) tile -> bf16 in LDS (2 rows per wave) ----
  for (int rr = 0; rr < 2; ++rr) {
    int r = wave * 2 + rr;
    const float4* xr = (const float4*)(x + (rowbase + r) * IN);
    float4 v[8];
    float s = 0.f, s2 = 0.f;
    for (int i = 0; i < 8; ++i) {
      v[i] = xr[i * 32 + lane];
      s  += v[i].x + v[i].y + v[i].z + v[i].w;
      s2 += v[i].x * v[i].x + v[i].y * v[i].y + v[i].z * v[i].z + v[i].w * v[i].w;
    }
    for (int off = 16; off; off >>= 1) { s += __shfl_xor(s, off, 32); s2 += __shfl_xor(s2, off, 32); }
    float mean = s * (1.f / 1024.f);
    float rstd = rsqrtf(s2 * (1.f / 1024.f) - mean * mean + 1e-5f);
    for (int i = 0; i < 8; ++i) {
      int c4 = i * 32 + lane;
      float4 g  = ((const float4*)ln_in_s)[c4];
      float4 be = ((const float4*)ln_in_b)[c4];
      __bf16* dst = shA + r * 1024 + c4 * 4;
      dst[0] = (__bf16)((v[i].x - mean) * rstd * g.x + be.x);
      dst[1] = (__bf16)((v[i].y - mean) * rstd * g.y + be.y);
      dst[2] = (__bf16)((v[i].z - mean) * rstd * g.z + be.z);
      dst[3] = (__bf16)((v[i].w - mean) * rstd * g.w + be.w);
    }
  }
  __syncthreads();

  const int arow = lane & 15;
  const int kbA  = (lane < 16) ? 0 : 16;    // A-frag K sub-base per half-wave
  const int colL = lane & 15;
  const int rbC  = (lane < 16) ? 0 : 8;     // C/D row base per half-wave

  // ---- step 2: q = A(16x1024) @ W_in -> 16x256 via WMMA bf16 ----
  {
    const v16bf* bp = (const v16bf*)WinF;
    for (int t = 0; t < 2; ++t) {
      int nt = wave * 2 + t;
      v8f acc = {};
      for (int ks = 0; ks < 32; ++ks) {
        const __bf16* ap = shA + arow * 1024 + ks * 32 + kbA;
        v8bf lo = *(const v8bf*)ap;
        v8bf hi = *(const v8bf*)(ap + 8);
        v16bf a = SHUF16(lo, hi);
        v16bf bf = bp[(nt * 32 + ks) * 32 + lane];
        acc = __builtin_amdgcn_wmma_f32_16x16x32_bf16(false, a, false, bf, (short)0, acc, false, false);
      }
      int col = nt * 16 + colL;
      for (int i = 0; i < 8; ++i) shq[(i + rbC) * 256 + col] = acc[i];
    }
  }
  __syncthreads();

  // ---- step 3: stage slot_memory as bf16 (batch-independent) ----
  for (int i = tid; i < NS * SD; i += 256) shSl[i] = (__bf16)slot_mem[i];
  __syncthreads();

  // ---- step 4: scores[r][h][n] = (q_h . k_h) * hd^-0.5 ----
  for (int i = 0; i < 16; ++i) {
    int id = i * 256 + tid;
    int r = id >> 8, h = (id >> 6) & 3, n = id & 63;
    const float*  qp = shq  + r * 256 + h * 64;
    const __bf16* kp = shSl + n * 256 + h * 64;
    float d = 0.f;
    for (int j = 0; j < 64; ++j) d += qp[j] * (float)kp[j];
    shsc[id] = d * 0.125f;
  }
  __syncthreads();

  // ---- step 5: softmax over 64 slots (one (r,h) pair per thread) ----
  if (tid < 64) {
    float* p = shsc + tid * 64;
    float mx = -1e30f;
    for (int n = 0; n < 64; ++n) mx = fmaxf(mx, p[n]);
    float sum = 0.f;
    for (int n = 0; n < 64; ++n) { float e = expf(p[n] - mx); p[n] = e; sum += e; }
    float inv = 1.f / sum;
    for (int n = 0; n < 64; ++n) p[n] *= inv;
  }
  __syncthreads();

  // ---- step 6: read[r][d] = sum_n attn[r][h(d)][n] * slots[n][d] ----
  float* shrd = shq;                         // overwrite q region
  for (int i = 0; i < 16; ++i) {
    int id = i * 256 + tid;
    int r = id >> 8, d = id & 255, h = d >> 6;
    const float*  ap = shsc + r * 256 + h * 64;
    const __bf16* sp = shSl + d;
    float acc = 0.f;
    for (int n = 0; n < 64; ++n) acc += ap[n] * (float)sp[n * 256];
    shrd[id] = acc;
  }
  __syncthreads();

  // ---- step 6b: read -> bf16; chunk summary (mean over 16 rows) ----
  for (int i = 0; i < 16; ++i) { int id = i * 256 + tid; shRb[id] = (__bf16)shrd[id]; }
  {
    float s = 0.f;
    for (int r = 0; r < 16; ++r) s += shrd[r * 256 + tid];
    shSum[tid] = s * (1.f / 16.f);
  }
  __syncthreads();

  // ---- step 6c: slot-state-independent chunk precompute ----
  const int cidx = b * NTILE + tile;
  float wvd = 0.f;                                   // wv = summary @ W_write
  for (int k = 0; k < 256; ++k) wvd += shSum[k] * W_write[k * 256 + tid];
  ws_wv[(size_t)cidx * 256 + tid] = wvd;
  float part = wvd * W_gate[256 + tid];              // wv half of gate logit
  for (int off = 16; off; off >>= 1) part += __shfl_xor(part, off, 32);
  if (lane == 0) shRed[wave] = part;
  if (tid < 64) {                                    // match (scale-free: only ordering matters)
    const float* kp = keyn + tid * 256;
    float d = 0.f;
    for (int k = 0; k < 256; ++k) d += shSum[k] * kp[k];
    shMt[tid] = d;
  }
  __syncthreads();
  if (tid == 0) {
    float z = 0.f;
    for (int w = 0; w < 8; ++w) z += shRed[w];
    ws_zwv[cidx] = z;
    int c0 = -1, c1 = -1;
    for (int j = 0; j < 3; ++j) {                    // top-3, first-index ties (lax.top_k)
      float best = -1e30f; int bi = 0;
      for (int n = 0; n < 64; ++n) {
        if (n == c0 || n == c1) continue;
        if (shMt[n] > best) { best = shMt[n]; bi = n; }
      }
      ws_top[cidx * 4 + j] = bi;
      if (j == 0) c0 = bi; else if (j == 1) c1 = bi;
    }
  }

  // ---- step 7: out = read(16x256) @ W_out -> 16x1024 via WMMA bf16 ----
  {
    float* op = out + rowbase * IN;
    const v16bf* bp = (const v16bf*)WoutF;
    for (int t = 0; t < 8; ++t) {
      int nt = wave * 8 + t;
      v8f acc = {};
      __builtin_prefetch(&bp[(nt * 8) * 32 + lane], 0, 0);
      for (int ks = 0; ks < 8; ++ks) {
        const __bf16* ap = shRb + arow * 256 + ks * 32 + kbA;
        v8bf lo = *(const v8bf*)ap;
        v8bf hi = *(const v8bf*)(ap + 8);
        v16bf a = SHUF16(lo, hi);
        v16bf bf = bp[(nt * 8 + ks) * 32 + lane];
        acc = __builtin_amdgcn_wmma_f32_16x16x32_bf16(false, a, false, bf, (short)0, acc, false, false);
      }
      int col = nt * 16 + colL;
      for (int i = 0; i < 8; ++i) op[(size_t)(i + rbC) * IN + col] = acc[i];
    }
  }
}

// ------------------------- phase 2: sequential scan ------------------------

__global__ __launch_bounds__(256)
void scan_slots(const float* __restrict__ slot_mem,
                const float* __restrict__ W_gate,
                const float* __restrict__ b_gate,
                const float* __restrict__ ln_s_s,
                const float* __restrict__ ln_s_b,
                const float* __restrict__ ws_wv,
                const float* __restrict__ ws_zwv,
                const int* __restrict__ ws_top,
                float* __restrict__ new_slots)
{
  __shared__ float slots[NS * SD];   // 64KB (gfx1250: up to 320KB/WG)
  __shared__ float red[8];
  const int tid = threadIdx.x, wave = tid >> 5, lane = tid & 31;
  const int b = blockIdx.x;

  for (int i = tid; i < NS * SD; i += 256) slots[i] = slot_mem[i];
  const float wg = W_gate[tid];
  const float bg = b_gate[0];
  const float* wvp = ws_wv + (size_t)b * NTILE * 256;
  const float* zp  = ws_zwv + b * NTILE;
  const int*   tp  = ws_top + b * NTILE * 4;
  __syncthreads();

  int i0 = tp[0], i1 = tp[1], i2 = tp[2];     // software-pipelined operands
  float wvd = wvp[tid], zwv = zp[0];
  for (int c = 0; c < NTILE; ++c) {
    int ni0 = 0, ni1 = 0, ni2 = 0; float nwv = 0.f, nz = 0.f;
    if (c + 1 < NTILE) {
      const int* tn = tp + (c + 1) * 4;
      ni0 = tn[0]; ni1 = tn[1]; ni2 = tn[2];
      nwv = wvp[(size_t)(c + 1) * 256 + tid];
      nz  = zp[c + 1];
    }
    float s0 = slots[i0 * 256 + tid];
    float s1 = slots[i1 * 256 + tid];
    float s2 = slots[i2 * 256 + tid];
    float sm = (s0 + s1 + s2) * (1.f / 3.f);
    float part = sm * wg;
    for (int off = 16; off; off >>= 1) part += __shfl_xor(part, off, 32);
    if (lane == 0) red[wave] = part;
    __syncthreads();
    float z = red[0] + red[1] + red[2] + red[3] + red[4] + red[5] + red[6] + red[7] + zwv + bg;
    float g = 1.f / (1.f + expf(-z));
    slots[i0 * 256 + tid] = (1.f - g) * s0 + g * wvd;
    slots[i1 * 256 + tid] = (1.f - g) * s1 + g * wvd;
    slots[i2 * 256 + tid] = (1.f - g) * s2 + g * wvd;
    __syncthreads();
    i0 = ni0; i1 = ni1; i2 = ni2; wvd = nwv; zwv = nz;
  }

  // final LayerNorm of slots -> new_slots (8 rows per wave)
  for (int rr = 0; rr < 8; ++rr) {
    int r = wave * 8 + rr;
    float vals[8], s = 0.f, s2 = 0.f;
    for (int j = 0; j < 8; ++j) {
      float v = slots[r * 256 + j * 32 + lane];
      vals[j] = v; s += v; s2 += v * v;
    }
    for (int off = 16; off; off >>= 1) { s += __shfl_xor(s, off, 32); s2 += __shfl_xor(s2, off, 32); }
    float mean = s * (1.f / 256.f);
    float rstd = rsqrtf(s2 * (1.f / 256.f) - mean * mean + 1e-5f);
    for (int j = 0; j < 8; ++j) {
      int d = j * 32 + lane;
      new_slots[((size_t)b * NS + r) * SD + d] = (vals[j] - mean) * rstd * ln_s_s[d] + ln_s_b[d];
    }
  }
}

// ------------------------------- launcher ----------------------------------

extern "C" void kernel_launch(void* const* d_in, const int* in_sizes, int n_in,
                              void* d_out, int out_size, void* d_ws, size_t ws_size,
                              hipStream_t stream) {
  const float* x        = (const float*)d_in[0];
  const float* slot_mem = (const float*)d_in[1];
  const float* slot_keys= (const float*)d_in[2];
  const float* W_in     = (const float*)d_in[3];
  const float* W_write  = (const float*)d_in[4];
  const float* W_gate   = (const float*)d_in[5];
  const float* b_gate   = (const float*)d_in[6];
  const float* W_out    = (const float*)d_in[7];
  const float* ln_in_s  = (const float*)d_in[8];
  const float* ln_in_b  = (const float*)d_in[9];
  const float* ln_s_s   = (const float*)d_in[10];
  const float* ln_s_b   = (const float*)d_in[11];

  // workspace layout (~3.2 MB total)
  char* ws = (char*)d_ws;
  __bf16* WinF  = (__bf16*)(ws);                 // 512 KB
  __bf16* WoutF = (__bf16*)(ws + 524288);        // 512 KB
  float*  keyn  = (float*)(ws + 1048576);        // 64 KB
  float*  wswv  = (float*)(ws + 1114112);        // 2 MB  (8*256 chunks * 256)
  float*  wszwv = (float*)(ws + 3211264);        // 8 KB
  int*    wstop = (int*)(ws + 3219456);          // 32 KB (stride 4)

  prep_win <<<1024, 256, 0, stream>>>(W_in, WinF);
  prep_wout<<<1024, 256, 0, stream>>>(W_out, WoutF);
  prep_keyn<<<64,   256, 0, stream>>>(slot_keys, keyn);

  float* out_main  = (float*)d_out;
  float* out_slots = out_main + (size_t)Bb * Ss * IN;

  dim3 g1(NTILE, Bb);
  fused_fwd<<<g1, 256, 0, stream>>>(x, slot_mem, W_write, W_gate, ln_in_s, ln_in_b,
                                    WinF, WoutF, keyn, out_main, wswv, wszwv, wstop);
  scan_slots<<<Bb, 256, 0, stream>>>(slot_mem, W_gate, b_gate, ln_s_s, ln_s_b,
                                     wswv, wszwv, wstop, out_slots);
}
